// MultiHeadAttention_63264868270429
// MI455X (gfx1250) — matrix operand order repaired and checked by reading, not compiled, not run on hardware
//
#include <hip/hip_runtime.h>
#include <stdint.h>

typedef _Float16 v16h __attribute__((ext_vector_type(16)));
typedef _Float16 v8h  __attribute__((ext_vector_type(8)));
typedef float    v8f  __attribute__((ext_vector_type(8)));
typedef unsigned int u32x4 __attribute__((ext_vector_type(4)));
typedef int          i32x4 __attribute__((ext_vector_type(4)));
typedef int          i32x8 __attribute__((ext_vector_type(8)));

#define NUM_HEADS 16
#define HIDDEN    1024
#define DK        64
#define BATCH     4
#define SEQ       2048
#define MROWS     (BATCH * SEQ)   // 8192

#if __has_builtin(__builtin_amdgcn_tensor_load_to_lds)
#define HAVE_TDM 1
#else
#define HAVE_TDM 0
#endif

// ---------------------------------------------------------------------------
// WMMA fragment loaders (wave32, v_wmma_f32_16x16x32_f16 A/B in this file's contiguous-k order, same for both)
// A (16x32 f16): lane l: row=l&15, g=l>>4; elem e -> K = e + 16g.
// B (32x16 f16): stored N-major with contiguous K; lane l: col=l&15, g=l>>4;
//                elem e -> K = e + 16g  (one contiguous v16h).
// C/D (16x16 f32): vgpr r, lane l: row = r + 8*(l>>4), col = l&15.
// ---------------------------------------------------------------------------
static __device__ inline v16h frag_a_load(const _Float16* base, int stride, int lane) {
    const int g = lane >> 4;
    const int r = lane & 15;
    const _Float16* p = base + r * stride + g * 16;
    v8h lo = *(const v8h*)(p);
    v8h hi = *(const v8h*)(p + 8);
    return __builtin_shufflevector(lo, hi, 0,1,2,3,4,5,6,7,8,9,10,11,12,13,14,15);
}

static __device__ inline v16h frag_b_load(const _Float16* base, int stride, int lane) {
    const int g = lane >> 4;
    const int c = lane & 15;
    return *(const v16h*)(base + c * stride + g * 16);
}

static __device__ inline v8f wmma_f16(v16h a, v16h b, v8f c) {
    return __builtin_amdgcn_wmma_f32_16x16x32_f16(false, a, false, b, (short)0, c,
                                                  false, false);
}

// ---------------------------------------------------------------------------
// TDM: DMA a 2D tile (rows x cols f16, row stride in elements) from global
// memory into LDS (packed contiguously: row y at lds + y*cols*2 bytes).
// Descriptor layout per CDNA5 ISA 8.3-8.5:
//   group0: count=1 | lds_addr | global_addr(57b) | type=2
//   group1: data_size=1(2B), tensor dims (huge => no OOB clamp),
//           tile_dim0=cols, tile_dim1=rows, tensor_dim0_stride=stride
// ---------------------------------------------------------------------------
#if HAVE_TDM
static __device__ inline void tdm_load_2d(const _Float16* lds_dst,
                                          const _Float16* gsrc,
                                          unsigned cols, unsigned rows,
                                          unsigned stride_elems) {
    unsigned long long ga = (unsigned long long)(uintptr_t)gsrc;
    unsigned lds_addr = (unsigned)(uintptr_t)lds_dst;   // low 32b = LDS byte addr

    u32x4 g0;
    g0[0] = 1u;                                       // count=1, user mode
    g0[1] = lds_addr;                                 // D#.lds_addr
    g0[2] = (unsigned)(ga & 0xFFFFFFFFu);             // global_addr[31:0]
    g0[3] = (unsigned)((ga >> 32) & 0x1FFFFFFu)       // global_addr[56:32]
          | (2u << 30);                               // type = 2 ("image")

    const unsigned td0 = 0x7FFFFFFFu;                 // huge dims: tile never OOB
    const unsigned td1 = 0x7FFFFFFFu;
    i32x8 g1;
    g1[0] = (int)(1u << 16);                          // data_size=1 (2 bytes)
    g1[1] = (int)((td0 & 0xFFFFu) << 16);             // tensor_dim0[15:0]
    g1[2] = (int)((td0 >> 16) | ((td1 & 0xFFFFu) << 16));
    g1[3] = (int)((td1 >> 16) | (cols << 16));        // tile_dim0 = cols
    g1[4] = (int)rows;                                // tile_dim1 = rows, tile_dim2=0
    g1[5] = (int)stride_elems;                        // tensor_dim0_stride[31:0]
    g1[6] = 0;
    g1[7] = 0;

    i32x4 z4 = {0, 0, 0, 0};
#if __has_include(<hip/amd_detail/amd_gfx1250_TDM.h>)
    i32x8 z8 = {0, 0, 0, 0, 0, 0, 0, 0};
    __builtin_amdgcn_tensor_load_to_lds(g0, g1, z4, z4, z8, 0);
#else
    __builtin_amdgcn_tensor_load_to_lds(g0, g1, z4, z4, 0);
#endif
}
#endif

// Issue staging of a tile; under TDM only wave 0 issues the DMA.
static __device__ inline void stage_issue(_Float16* lds_dst, const _Float16* gsrc,
                                          unsigned cols, unsigned rows,
                                          unsigned stride_elems, int wave) {
#if HAVE_TDM
    if (wave == 0) tdm_load_2d(lds_dst, gsrc, cols, rows, stride_elems);
#else
    // cooperative copy: 256 threads, v8h chunks
    const int tid = threadIdx.x;
    const int cpr = cols >> 3;                        // chunks per row
    const int total = (int)rows * cpr;
    for (int i = tid; i < total; i += 256) {
        int r = i / cpr, c = (i - r * cpr) << 3;
        *(v8h*)(lds_dst + r * cols + c) = *(const v8h*)(gsrc + (size_t)r * stride_elems + c);
    }
#endif
}

static __device__ inline void stage_wait(int wave) {
#if HAVE_TDM
    if (wave == 0) __builtin_amdgcn_s_wait_tensorcnt(0);
#else
    (void)wave;
#endif
}

// ---------------------------------------------------------------------------
// fp32 -> f16 conversion (grid-stride)
// ---------------------------------------------------------------------------
__global__ void cvt_f32_f16(const float* __restrict__ in, _Float16* __restrict__ out,
                            int n) {
    int i = blockIdx.x * blockDim.x + threadIdx.x;
    int stride = gridDim.x * blockDim.x;
    for (; i < n; i += stride) out[i] = (_Float16)in[i];
}

// ---------------------------------------------------------------------------
// Projection GEMM:  out[m,n] = sum_k X[m,k] * W[n,k] + bias[n]
// M=8192, N=1024, K=1024. Block=256thr (8 waves), block tile 128x128,
// wave tile 32x64. A(128x32) and B(128x32) tiles TDM-staged in LDS,
// double-buffered: issue k+32 DMA while computing on k.
// transposed==0: out[((b*16+h)*SEQ+l)*DK+d] (Q,K);  ==1: [.. *DK+d)*SEQ+l] (V^T)
// ---------------------------------------------------------------------------
__global__ void proj_kernel(const _Float16* __restrict__ X,
                            const _Float16* __restrict__ W,
                            const float* __restrict__ bias,
                            _Float16* __restrict__ outp,
                            int transposed) {
    const int lane = threadIdx.x & 31;
    const int wave = threadIdx.x >> 5;
    const int g    = lane >> 4;
    const int wm   = wave & 3;                       // 4 waves along M
    const int wn   = wave >> 2;                      // 2 waves along N
    const int m_blk = blockIdx.x * 128;
    const int n_blk = blockIdx.y * 128;
    const int m_wave = m_blk + wm * 32;
    const int n_loc  = wn * 64;                      // wave's N base within block
    const int n_wave = n_blk + n_loc;

    __shared__ _Float16 sA[2][128 * 32];             // 8 KB x2
    __shared__ _Float16 sB[2][128 * 32];             // 8 KB x2

    v8f acc[2][4];
    #pragma unroll
    for (int mi = 0; mi < 2; ++mi)
        #pragma unroll
        for (int ni = 0; ni < 4; ++ni) acc[mi][ni] = (v8f){};

    float bv[4];
    #pragma unroll
    for (int ni = 0; ni < 4; ++ni)
        bv[ni] = bias[n_wave + ni * 16 + (lane & 15)];

    // prologue: stage k-step 0
    stage_issue(sA[0], X + (size_t)m_blk * HIDDEN, 32, 128, HIDDEN, wave);
    stage_issue(sB[0], W + (size_t)n_blk * HIDDEN, 32, 128, HIDDEN, wave);

    int buf = 0;
    for (int kk = 0; kk < HIDDEN; kk += 32) {
        stage_wait(wave);
        __syncthreads();                             // staged tiles visible
        if (kk + 32 < HIDDEN) {                      // prefetch next k-step
            stage_issue(sA[buf ^ 1], X + (size_t)m_blk * HIDDEN + kk + 32,
                        32, 128, HIDDEN, wave);
            stage_issue(sB[buf ^ 1], W + (size_t)n_blk * HIDDEN + kk + 32,
                        32, 128, HIDDEN, wave);
        }
        const _Float16* At = sA[buf];
        const _Float16* Bt = sB[buf];

        v16h a0 = frag_a_load(At + (wm * 32 +  0) * 32, 32, lane);
        v16h a1 = frag_a_load(At + (wm * 32 + 16) * 32, 32, lane);
        v16h b0 = frag_b_load(Bt + (n_loc +  0) * 32, 32, lane);
        v16h b1 = frag_b_load(Bt + (n_loc + 16) * 32, 32, lane);
        v16h b2 = frag_b_load(Bt + (n_loc + 32) * 32, 32, lane);
        v16h b3 = frag_b_load(Bt + (n_loc + 48) * 32, 32, lane);
        acc[0][0] = wmma_f16(a0, b0, acc[0][0]);
        acc[0][1] = wmma_f16(a0, b1, acc[0][1]);
        acc[0][2] = wmma_f16(a0, b2, acc[0][2]);
        acc[0][3] = wmma_f16(a0, b3, acc[0][3]);
        acc[1][0] = wmma_f16(a1, b0, acc[1][0]);
        acc[1][1] = wmma_f16(a1, b1, acc[1][1]);
        acc[1][2] = wmma_f16(a1, b2, acc[1][2]);
        acc[1][3] = wmma_f16(a1, b3, acc[1][3]);

        buf ^= 1;
        __syncthreads();                             // all reads retired before
    }                                                // this buffer is re-targeted

    #pragma unroll
    for (int mi = 0; mi < 2; ++mi) {
        #pragma unroll
        for (int ni = 0; ni < 4; ++ni) {
            #pragma unroll
            for (int r = 0; r < 8; ++r) {
                float val = acc[mi][ni][r] + bv[ni];
                int Mg = m_wave + mi * 16 + r + 8 * g;   // b*SEQ + l
                int n  = n_wave + ni * 16 + (lane & 15); // h*DK + d
                int b  = Mg >> 11;
                int l  = Mg & (SEQ - 1);
                int h  = n >> 6;
                int d  = n & (DK - 1);
                if (transposed)
                    outp[(((size_t)(b * NUM_HEADS + h)) * DK + d) * SEQ + l] = (_Float16)val;
                else
                    outp[(((size_t)(b * NUM_HEADS + h)) * SEQ + l) * DK + d] = (_Float16)val;
            }
        }
    }
}

// ---------------------------------------------------------------------------
// Flash-attention: one block per (b*16+h, 128-row query tile), 8 waves,
// each wave owns 16 query rows. K (32x64) and V^T (64x32) tiles are shared by
// all 8 waves -> TDM-staged in LDS (double-buffered) to kill the 8x redundant
// global traffic. Online softmax; P repacked C-layout -> A-layout via LDS.
// ---------------------------------------------------------------------------
__global__ void attn_kernel(const _Float16* __restrict__ Qh,   // [BH, SEQ, DK]
                            const _Float16* __restrict__ Kh,   // [BH, SEQ, DK]
                            const _Float16* __restrict__ Vth,  // [BH, DK, SEQ]
                            float* __restrict__ out) {         // [BH, SEQ, DK] f32
    const int lane = threadIdx.x & 31;
    const int wave = threadIdx.x >> 5;
    const int g    = lane >> 4;
    const int bh   = blockIdx.x;                    // 0..63
    const int l0   = blockIdx.y * 128 + wave * 16;  // query-row base of this wave

    __shared__ _Float16 sK[2][32 * 64];             // 4 KB x2 (row t_rel, d contig)
    __shared__ _Float16 sV[2][64 * 32];             // 4 KB x2 (row d, t_rel contig)
    __shared__ _Float16 lds_p[8 * 16 * 40];         // per-wave P repack scratch
    _Float16* pbuf = &lds_p[wave * 16 * 40];

    const _Float16* Qb = Qh  + (size_t)bh * SEQ * DK;
    const _Float16* Kb = Kh  + (size_t)bh * SEQ * DK;
    const _Float16* Vb = Vth + (size_t)bh * DK * SEQ;

    // Q fragments: 16 rows x 64 d -> two K=32 A-frags (held in VGPRs)
    v16h aq0 = frag_a_load(Qb + (size_t)l0 * DK + 0,  DK, lane);
    v16h aq1 = frag_a_load(Qb + (size_t)l0 * DK + 32, DK, lane);

    v8f o0 = (v8f){}, o1 = (v8f){}, o2 = (v8f){}, o3 = (v8f){};
    float mrun[8], lrun[8];
    #pragma unroll
    for (int r = 0; r < 8; ++r) { mrun[r] = -1e30f; lrun[r] = 0.0f; }

    // prologue: stage t-step 0
    stage_issue(sK[0], Kb, 64, 32, DK, wave);
    stage_issue(sV[0], Vb, 32, 64, SEQ, wave);

    int buf = 0;
    for (int t0 = 0; t0 < SEQ; t0 += 32) {
        stage_wait(wave);
        __syncthreads();
        if (t0 + 32 < SEQ) {                        // prefetch next t-step
            stage_issue(sK[buf ^ 1], Kb + (size_t)(t0 + 32) * DK, 64, 32, DK, wave);
            stage_issue(sV[buf ^ 1], Vb + (t0 + 32), 32, 64, SEQ, wave);
        }
        const _Float16* Kt = sK[buf];
        const _Float16* Vt = sV[buf];

        // ---- scores: two 16x16 tiles of Q K^T (K-dim = d, split 2x32) ----
        v16h bk00 = frag_b_load(Kt +  0 * 64 +  0, 64, lane);
        v16h bk01 = frag_b_load(Kt +  0 * 64 + 32, 64, lane);
        v16h bk10 = frag_b_load(Kt + 16 * 64 +  0, 64, lane);
        v16h bk11 = frag_b_load(Kt + 16 * 64 + 32, 64, lane);
        v8f s0 = (v8f){}, s1 = (v8f){};
        s0 = wmma_f16(aq0, bk00, s0);
        s0 = wmma_f16(aq1, bk01, s0);
        s1 = wmma_f16(aq0, bk10, s1);
        s1 = wmma_f16(aq1, bk11, s1);

        // ---- online softmax (row m = r + 8*g lives in 16 lanes of a half) ----
        #pragma unroll
        for (int r = 0; r < 8; ++r) {
            float v0 = s0[r] * 0.125f;              // 1/sqrt(64)
            float v1 = s1[r] * 0.125f;
            float rm = fmaxf(v0, v1);
            rm = fmaxf(rm, __shfl_xor(rm, 1));
            rm = fmaxf(rm, __shfl_xor(rm, 2));
            rm = fmaxf(rm, __shfl_xor(rm, 4));
            rm = fmaxf(rm, __shfl_xor(rm, 8));
            float mnew  = fmaxf(mrun[r], rm);
            float alpha = __expf(mrun[r] - mnew);
            float p0    = __expf(v0 - mnew);
            float p1    = __expf(v1 - mnew);
            float rs    = p0 + p1;
            rs += __shfl_xor(rs, 1);
            rs += __shfl_xor(rs, 2);
            rs += __shfl_xor(rs, 4);
            rs += __shfl_xor(rs, 8);
            lrun[r] = lrun[r] * alpha + rs;
            mrun[r] = mnew;
            o0[r] *= alpha; o1[r] *= alpha; o2[r] *= alpha; o3[r] *= alpha;

            int m = r + 8 * g;                      // stash P (C-layout) to LDS
            pbuf[m * 40 +      (lane & 15)] = (_Float16)p0;
            pbuf[m * 40 + 16 + (lane & 15)] = (_Float16)p1;
        }

        // same-wave LDS ops are in-order (ISA 7.3): repack to A-fragment
        v16h pa = frag_a_load(pbuf, 40, lane);

        // ---- O += P @ V  (K-dim = t, 32) ----
        v16h bv0 = frag_b_load(Vt +  0 * 32, 32, lane);
        v16h bv1 = frag_b_load(Vt + 16 * 32, 32, lane);
        v16h bv2 = frag_b_load(Vt + 32 * 32, 32, lane);
        v16h bv3 = frag_b_load(Vt + 48 * 32, 32, lane);
        o0 = wmma_f16(pa, bv0, o0);
        o1 = wmma_f16(pa, bv1, o1);
        o2 = wmma_f16(pa, bv2, o2);
        o3 = wmma_f16(pa, bv3, o3);

        buf ^= 1;
        __syncthreads();
    }

    // ---- epilogue: normalize and write fp32 output (b,h,l,d order) ----
    #pragma unroll
    for (int r = 0; r < 8; ++r) {
        float inv = 1.0f / lrun[r];
        int l = l0 + r + 8 * g;
        float* orow = out + ((size_t)bh * SEQ + l) * DK;
        orow[ 0 + (lane & 15)] = o0[r] * inv;
        orow[16 + (lane & 15)] = o1[r] * inv;
        orow[32 + (lane & 15)] = o2[r] * inv;
        orow[48 + (lane & 15)] = o3[r] * inv;
    }
}

// ---------------------------------------------------------------------------
// Host-side launcher
// inputs: x, Wq, bq, Wk, bk, Wv, bv  (all fp32); output [B,H,L,DK] fp32
// ---------------------------------------------------------------------------
extern "C" void kernel_launch(void* const* d_in, const int* in_sizes, int n_in,
                              void* d_out, int out_size, void* d_ws, size_t ws_size,
                              hipStream_t stream) {
    (void)in_sizes; (void)n_in; (void)out_size; (void)ws_size;

    const float* x  = (const float*)d_in[0];
    const float* Wq = (const float*)d_in[1];
    const float* bq = (const float*)d_in[2];
    const float* Wk = (const float*)d_in[3];
    const float* bk = (const float*)d_in[4];
    const float* Wv = (const float*)d_in[5];
    const float* bv = (const float*)d_in[6];
    float* out = (float*)d_out;

    const size_t NX = (size_t)MROWS * HIDDEN;   // 8.39M elems
    const size_t NW = (size_t)HIDDEN * HIDDEN;  // 1.05M elems

    _Float16* xh  = (_Float16*)d_ws;
    _Float16* wqh = xh  + NX;
    _Float16* wkh = wqh + NW;
    _Float16* wvh = wkh + NW;
    _Float16* Qh  = wvh + NW;                   // [BH, SEQ, DK]
    _Float16* Kh  = Qh  + NX;                   // [BH, SEQ, DK]
    _Float16* Vth = Kh  + NX;                   // [BH, DK, SEQ]

    cvt_f32_f16<<<2048, 256, 0, stream>>>(x,  xh,  (int)NX);
    cvt_f32_f16<<<1024, 256, 0, stream>>>(Wq, wqh, (int)NW);
    cvt_f32_f16<<<1024, 256, 0, stream>>>(Wk, wkh, (int)NW);
    cvt_f32_f16<<<1024, 256, 0, stream>>>(Wv, wvh, (int)NW);

    dim3 pg(MROWS / 128, HIDDEN / 128);         // 64 x 8
    proj_kernel<<<pg, 256, 0, stream>>>(xh, wqh, bq, Qh, 0);
    proj_kernel<<<pg, 256, 0, stream>>>(xh, wkh, bk, Kh, 0);
    proj_kernel<<<pg, 256, 0, stream>>>(xh, wvh, bv, Vth, 1);

    dim3 ag(BATCH * NUM_HEADS, SEQ / 128);      // 64 x 16
    attn_kernel<<<ag, 256, 0, stream>>>(Qh, Kh, Vth, out);
}
